// LaplaceAttention_51316269253360
// MI455X (gfx1250) — matrix operand order repaired and checked by reading, not compiled, not run on hardware
//
#include <hip/hip_runtime.h>
#include <hip/hip_bf16.h>

// Problem constants (from reference): N=2, C=512, S=256, SCALE=2
#define N_    2
#define C_    512
#define S_    256
#define RT    16          // q-row tile per block
#define DCH   16          // k rows per TDM chunk
#define NCH   (C_ / DCH)  // 32 chunks
#define NTHR  256         // 8 wave32s

typedef __attribute__((ext_vector_type(16))) _Float16     v16h;
typedef __attribute__((ext_vector_type(8)))  float        v8f;
typedef __attribute__((ext_vector_type(4)))  unsigned int v4u;
typedef __attribute__((ext_vector_type(8)))  int          v8i;
typedef __attribute__((ext_vector_type(4)))  int          v4i;

// ---------------------------------------------------------------------------
// TDM: issue TENSOR_LOAD_TO_LDS of a DCH x S_ f32 tile (row stride S_) from
// global `gsrc` into LDS byte offset `lds_byte_off`.
// D# built per CDNA5 ISA §8.3/§8.4 (group0: count/lds/global/type,
// group1: data_size=4B, tensor_dim0=S, tensor_dim1=C, tile=256x16, stride=S).
// 6-arg builtin form (clang-23 / therock-10.0 headers).
// ---------------------------------------------------------------------------
__device__ __forceinline__ void tdm_load_k_chunk(const float* gsrc,
                                                 unsigned lds_byte_off) {
  unsigned long long ga = (unsigned long long)(uintptr_t)gsrc;
  v4u g0;
  g0[0] = 1u;                                   // count=1, is_restore=0, no gather
  g0[1] = lds_byte_off;                         // lds_addr
  g0[2] = (unsigned)ga;                         // global_addr[31:0]
  g0[3] = ((unsigned)(ga >> 32) & 0x01FFFFFFu)  // global_addr[56:32]
        | (2u << 30);                           // type=2 ("image")
  v8i g1;
  g1[0] = 0x00020000;                           // wg_mask=0, data_size=2 (4B)
  g1[1] = (int)(256u << 16);                    // tensor_dim0 = 256 (S)
  g1[2] = (int)(512u << 16);                    // tensor_dim1 = 512 (C)
  g1[3] = (int)(256u << 16);                    // tile_dim0   = 256
  g1[4] = DCH;                                  // tile_dim1 = 16, tile_dim2 = 0
  g1[5] = 256;                                  // tensor_dim0_stride = S (lo32)
  g1[6] = 0;                                    // stride hi16 / dim1_stride lo
  g1[7] = 0;
  v4i g2 = {0, 0, 0, 0};                        // 2D tile: dims 2/3 unused
  v4i g3 = {0, 0, 0, 0};
  v8i g4 = {0, 0, 0, 0, 0, 0, 0, 0};            // extra group (clang-23 form)
  __builtin_amdgcn_tensor_load_to_lds(g0, g1, g2, g3, g4, 0);
}

__global__ __launch_bounds__(NTHR)
void laplace_attn_kernel(const float* __restrict__ q,
                         const float* __restrict__ k,
                         const float* __restrict__ v,
                         float* __restrict__ out) {
  // LDS: [0,32K) k double buffer | [32K,64K) f32 scores | [64K,80K) f16 weights
  __shared__ unsigned char smem[2 * DCH * S_ * 4 + RT * C_ * 4 + RT * C_ * 2]
      __attribute__((aligned(16)));
  float*    kbuf = (float*)smem;
  float*    sc   = (float*)(smem + 2 * DCH * S_ * 4);
  _Float16* w16  = (_Float16*)(smem + 2 * DCH * S_ * 4 + RT * C_ * 4);

  const int tid  = threadIdx.x;
  const int w    = tid >> 5;         // wave id 0..7
  const int l    = tid & 31;         // lane id
  const int n    = blockIdx.x >> 5;  // batch
  const int tile = blockIdx.x & 31;  // q-row tile
  const int c0   = tile * RT;

  const float* qn = q + (size_t)n * C_ * S_;
  const float* kn = k + (size_t)n * C_ * S_;
  const float* vn = v + (size_t)n * C_ * S_;
  float*       on = out + (size_t)n * C_ * S_;

  const unsigned smem_lo = (unsigned)(uintptr_t)(void*)smem;  // LDS byte offset

  // ---- pin this wave's two q rows in VGPRs (lane l holds s = l + 32j) ----
  const int r0 = 2 * w, r1 = 2 * w + 1;
  float q0r[8], q1r[8];
#pragma unroll
  for (int j = 0; j < 8; ++j) {
    q0r[j] = qn[(size_t)(c0 + r0) * S_ + l + 32 * j];
    q1r[j] = qn[(size_t)(c0 + r1) * S_ + l + 32 * j];
  }

  // ---- phase 1: L1-distance scores, k streamed via TDM double buffer ----
  if (w == 0) {
    tdm_load_k_chunk(kn, smem_lo);  // chunk 0 -> buf 0
  }
  for (int ch = 0; ch < NCH; ++ch) {
    __syncthreads();  // everyone done reading buf[(ch+1)&1] from prev iter
    if (w == 0) {
      if (ch + 1 < NCH) {
        tdm_load_k_chunk(kn + (size_t)(ch + 1) * DCH * S_,
                         smem_lo + (unsigned)(((ch + 1) & 1) * DCH * S_ * 4));
        __builtin_amdgcn_s_wait_tensorcnt(1);  // chunk ch complete (in-order)
      } else {
        __builtin_amdgcn_s_wait_tensorcnt(0);
      }
    }
    __syncthreads();  // publish buf[ch&1]
    const float* kb = kbuf + (ch & 1) * DCH * S_;
    for (int dl = 0; dl < DCH; ++dl) {
      const float* kr = kb + dl * S_;
      float s0 = 0.f, s1 = 0.f;
#pragma unroll
      for (int j = 0; j < 8; ++j) {
        float kv = kr[l + 32 * j];                 // ds_load_b32, conflict-free
        s0 += __builtin_fabsf(kv - q0r[j]);        // v_add_f32 ..., |diff|
        s1 += __builtin_fabsf(kv - q1r[j]);
      }
#pragma unroll
      for (int m = 16; m >= 1; m >>= 1) {          // wave32 tree reduction
        s0 += __shfl_xor(s0, m, 32);
        s1 += __shfl_xor(s1, m, 32);
      }
      if (l == 0) {
        int d = ch * DCH + dl;
        sc[r0 * C_ + d] = -0.5f * s0;              // -SAD/SCALE
        sc[r1 * C_ + d] = -0.5f * s1;
      }
    }
  }
  __syncthreads();

  // ---- phase 2: exact softmax per row, emit f16 weights ----
#pragma unroll
  for (int rr = 0; rr < 2; ++rr) {
    const int r = 2 * w + rr;
    const float* srow = sc + r * C_;
    float x[16];
#pragma unroll
    for (int j = 0; j < 16; ++j) x[j] = srow[l + 32 * j];
    float mx = x[0];
#pragma unroll
    for (int j = 1; j < 16; ++j) mx = fmaxf(mx, x[j]);
#pragma unroll
    for (int m = 16; m >= 1; m >>= 1) mx = fmaxf(mx, __shfl_xor(mx, m, 32));
    float sum = 0.f;
#pragma unroll
    for (int j = 0; j < 16; ++j) {
      x[j] = __expf(x[j] - mx);                    // v_exp_f32
      sum += x[j];
    }
#pragma unroll
    for (int m = 16; m >= 1; m >>= 1) sum += __shfl_xor(sum, m, 32);
    const float inv = 1.0f / sum;
    _Float16* wrow = w16 + r * C_;
#pragma unroll
    for (int j = 0; j < 16; ++j)
      wrow[l + 32 * j] = (_Float16)(x[j] * inv);
  }
  __syncthreads();

  // ---- phase 3: rep = W @ V via v_wmma_f32_16x16x32_f16 ----
  // Wave w owns output columns [32w, 32w+32): two 16x16 accumulator tiles.
  const int half = l >> 4;   // lane half-wave
  const int lm   = l & 15;
  const int cb0  = w * 32;
  const int cb1  = w * 32 + 16;
  v8f acc0 = {};
  v8f acc1 = {};
  for (int k0 = 0; k0 < C_; k0 += 32) {
    // A operand: 16x32 f16 tile of W. ISA layout: M = lane%16; VGPR t holds
    // K = {g*16 + half*8 + 2u, +1} with g=t>>2, u=t&3 (pairs are dword-packed).
    union { v16h h; unsigned int u32[8]; } A;
    const unsigned int* arow = (const unsigned int*)(w16 + lm * C_);
#pragma unroll
    for (int t = 0; t < 8; ++t) {
      const int g = t >> 2, u = t & 3;
      A.u32[t] = arow[(k0 >> 1) + g * 8 + half * 4 + u];
    }
    // B operand: 32x16 f16 tile of V. ISA layout: N = lane%16; VGPR j holds
    // K = {8*half + 2j + 8*(j>>2), +1}. Convert f32 -> f16 on load.
    union { v16h h; _Float16 e[16]; } B0, B1;
#pragma unroll
    for (int j = 0; j < 8; ++j) {
      const int d = k0 + 8 * half + 2 * j + 8 * (j >> 2);
      B0.e[2 * j]     = (_Float16)vn[(size_t)d * S_ + cb0 + lm];
      B0.e[2 * j + 1] = (_Float16)vn[(size_t)(d + 1) * S_ + cb0 + lm];
      B1.e[2 * j]     = (_Float16)vn[(size_t)d * S_ + cb1 + lm];
      B1.e[2 * j + 1] = (_Float16)vn[(size_t)(d + 1) * S_ + cb1 + lm];
    }
    acc0 = __builtin_amdgcn_wmma_f32_16x16x32_f16(
        false, A.h, false, B0.h, (short)0, acc0, false, false);
    acc1 = __builtin_amdgcn_wmma_f32_16x16x32_f16(
        false, A.h, false, B1.h, (short)0, acc1, false, false);
  }
  // D layout: c[r] = D[M = r + 8*half][N = lane%16]
#pragma unroll
  for (int r = 0; r < 8; ++r) {
    on[(size_t)(c0 + r + 8 * half) * S_ + cb0 + lm] = acc0[r];
    on[(size_t)(c0 + r + 8 * half) * S_ + cb1 + lm] = acc1[r];
  }
}

extern "C" void kernel_launch(void* const* d_in, const int* in_sizes, int n_in,
                              void* d_out, int out_size, void* d_ws,
                              size_t ws_size, hipStream_t stream) {
  (void)in_sizes; (void)n_in; (void)out_size; (void)d_ws; (void)ws_size;
  const float* q = (const float*)d_in[0];
  const float* k = (const float*)d_in[1];
  const float* v = (const float*)d_in[2];
  float* out = (float*)d_out;
  dim3 grid(N_ * (C_ / RT));  // 64 blocks
  dim3 block(NTHR);           // 8 wave32s
  hipLaunchKernelGGL(laplace_attn_kernel, grid, block, 0, stream, q, k, v, out);
}
